// GATSPMVConv_77953656422439
// MI455X (gfx1250) — matrix operand order repaired and checked by reading, not compiled, not run on hardware
//
#include <hip/hip_runtime.h>
#include <hip/hip_fp16.h>
#include <math.h>

// GAT layer: N=100000 nodes, IN=128, H=4 heads, C=12, DEG=16 contiguous in-edges/node.
#define GAT_IN   128
#define GAT_H    4
#define GAT_C    12
#define GAT_HC   48
#define GAT_DEG  16
#define NEG_SLOPE 0.2f
#define WT_PITCH 144   // padded LDS row in halfs: 288B -> 32B-aligned v16h loads, bank-skewed

typedef __attribute__((ext_vector_type(16))) _Float16 v16h;
typedef __attribute__((ext_vector_type(8)))  float    v8f;

// ---------------------------------------------------------------------------
// Kernel 1: h = x @ W via split-f16 WMMA (near-fp32 accuracy, f32 accumulate).
// One wave computes a 16-node x 48-channel tile. Weight staged in LDS (hi/lo).
// ---------------------------------------------------------------------------
__global__ __launch_bounds__(256)
void gat_gemm_wmma(const float* __restrict__ x, const float* __restrict__ w,
                   float* __restrict__ h, int nNodes) {
  __shared__ _Float16 wHi[GAT_HC * WT_PITCH];
  __shared__ _Float16 wLo[GAT_HC * WT_PITCH];

  // Cooperative transpose + hi/lo split of W[128][48] -> wT[48][128] (f16).
  for (int idx = threadIdx.x; idx < GAT_HC * GAT_IN; idx += blockDim.x) {
    int n = idx >> 7;          // output channel 0..47
    int k = idx & 127;         // input channel 0..127
    float v = w[k * GAT_HC + n];
    _Float16 hi = (_Float16)v;
    _Float16 lo = (_Float16)(v - (float)hi);
    wHi[n * WT_PITCH + k] = hi;
    wLo[n * WT_PITCH + k] = lo;
  }
  __syncthreads();

  int lane = threadIdx.x & 31;
  int wave = blockIdx.x * (blockDim.x >> 5) + (threadIdx.x >> 5);
  int m0 = wave * 16;                       // first node of this wave's tile
  if (m0 >= nNodes) return;

  int mrow = m0 + (lane & 15);
  if (mrow >= nNodes) mrow = nNodes - 1;    // safe duplicate read (N%16==0 anyway)
  const float* xrow = x + (size_t)mrow * GAT_IN;

  v8f acc[3];
  #pragma unroll
  for (int nt = 0; nt < 3; ++nt) { v8f z = {}; acc[nt] = z; }

  const int aBase = (lane & 16) ? 16 : 0;   // A frag: K(i) = aBase + i
  const int bBase = (lane & 16) ? 16 : 0;   // B frag: K(i) = bBase + i
  const int ncol  = lane & 15;

  #pragma unroll
  for (int kc = 0; kc < 4; ++kc) {          // K = 128 in chunks of 32
    const float* ap = xrow + kc * 32 + aBase;
    float4 f0 = *(const float4*)(ap +  0);
    float4 f1 = *(const float4*)(ap +  4);
    float4 f2 = *(const float4*)(ap +  8);
    float4 f3 = *(const float4*)(ap + 12);
    float af[16] = {f0.x,f0.y,f0.z,f0.w, f1.x,f1.y,f1.z,f1.w,
                    f2.x,f2.y,f2.z,f2.w, f3.x,f3.y,f3.z,f3.w};
    v16h ahi, alo;
    #pragma unroll
    for (int i = 0; i < 16; ++i) {
      _Float16 hi = (_Float16)af[i];
      ahi[i] = hi;
      alo[i] = (_Float16)(af[i] - (float)hi);
    }
    int kb = kc * 32 + bBase;
    #pragma unroll
    for (int nt = 0; nt < 3; ++nt) {
      v16h bhi = *(const v16h*)(&wHi[(nt * 16 + ncol) * WT_PITCH + kb]);
      v16h blo = *(const v16h*)(&wLo[(nt * 16 + ncol) * WT_PITCH + kb]);
      acc[nt] = __builtin_amdgcn_wmma_f32_16x16x32_f16(false, ahi, false, bhi,
                                                       (short)0, acc[nt], false, false);
      acc[nt] = __builtin_amdgcn_wmma_f32_16x16x32_f16(false, ahi, false, blo,
                                                       (short)0, acc[nt], false, false);
      acc[nt] = __builtin_amdgcn_wmma_f32_16x16x32_f16(false, alo, false, bhi,
                                                       (short)0, acc[nt], false, false);
    }
  }

  // D layout: VGPR r -> M = r + (lane<16?0:8), N = lane%16.
  int moff = (lane & 16) ? 8 : 0;
  #pragma unroll
  for (int nt = 0; nt < 3; ++nt) {
    int col = nt * 16 + ncol;
    #pragma unroll
    for (int r = 0; r < 8; ++r) {
      int node = m0 + moff + r;
      if (node < nNodes) h[(size_t)node * GAT_HC + col] = acc[nt][r];
    }
  }
}

// ---------------------------------------------------------------------------
// Kernel 2: per-node attention logit halves a1/a2 = <h[n,hd,:], att_{l,r}[hd,:]>
// ---------------------------------------------------------------------------
__global__ __launch_bounds__(256)
void gat_logits(const float* __restrict__ h, const float* __restrict__ att_l,
                const float* __restrict__ att_r, float* __restrict__ a1,
                float* __restrict__ a2, int total) {
  int t = blockIdx.x * blockDim.x + threadIdx.x;
  if (t >= total) return;
  int hd = t & (GAT_H - 1);
  const float* hp = h + (size_t)(t >> 2) * GAT_HC + hd * GAT_C;
  float s1 = 0.f, s2 = 0.f;
  #pragma unroll
  for (int c = 0; c < GAT_C; ++c) {
    float v = hp[c];
    s1 += v * att_l[hd * GAT_C + c];
    s2 += v * att_r[hd * GAT_C + c];
  }
  a1[t] = s1;
  a2[t] = s2;
}

// ---------------------------------------------------------------------------
// Kernel 3: edge softmax (16 contiguous in-edges per node) + weighted gather.
// One thread per (node, head).
// ---------------------------------------------------------------------------
__global__ __launch_bounds__(256)
void gat_softmax_agg(const float* __restrict__ h, const float* __restrict__ a1,
                     const float* __restrict__ a2, const int* __restrict__ src,
                     float* __restrict__ out, int total) {
  int t = blockIdx.x * blockDim.x + threadIdx.x;
  if (t >= total) return;
  int hd = t & (GAT_H - 1);
  int n  = t >> 2;
  float a2n = a2[t];

  int   s[GAT_DEG];
  float ev[GAT_DEG];
  float m = -1e30f;
  #pragma unroll
  for (int j = 0; j < GAT_DEG; ++j) {
    int sj = src[n * GAT_DEG + j];
    s[j] = sj;
    float e = a1[sj * GAT_H + hd] + a2n;
    e = (e > 0.f) ? e : NEG_SLOPE * e;         // leaky relu
    ev[j] = e;
    m = fmaxf(m, e);
  }
  float z = 0.f;
  #pragma unroll
  for (int j = 0; j < GAT_DEG; ++j) {
    float a = expf(ev[j] - m);                 // stable softmax
    ev[j] = a;
    z += a;
  }
  float acc[GAT_C] = {};
  #pragma unroll
  for (int j = 0; j < GAT_DEG; ++j) {
    const float4* hp = (const float4*)(h + (size_t)s[j] * GAT_HC + hd * GAT_C);
    float wj = ev[j];
    float4 p0 = hp[0], p1 = hp[1], p2 = hp[2];
    acc[0] += wj * p0.x; acc[1]  += wj * p0.y; acc[2]  += wj * p0.z; acc[3]  += wj * p0.w;
    acc[4] += wj * p1.x; acc[5]  += wj * p1.y; acc[6]  += wj * p1.z; acc[7]  += wj * p1.w;
    acc[8] += wj * p2.x; acc[9]  += wj * p2.y; acc[10] += wj * p2.z; acc[11] += wj * p2.w;
  }
  float invz = 1.f / z;
  float* op = out + (size_t)n * GAT_HC + hd * GAT_C;
  #pragma unroll
  for (int c = 0; c < GAT_C; ++c) op[c] = acc[c] * invz;
}

// ---------------------------------------------------------------------------
extern "C" void kernel_launch(void* const* d_in, const int* in_sizes, int n_in,
                              void* d_out, int out_size, void* d_ws, size_t ws_size,
                              hipStream_t stream) {
  const float* x     = (const float*)d_in[0];   // [N,128]
  const float* wgt   = (const float*)d_in[1];   // [128,48]
  const float* att_l = (const float*)d_in[2];   // [4,12]
  const float* att_r = (const float*)d_in[3];   // [4,12]
  const int*   src   = (const int*)d_in[4];     // [E]
  // d_in[5] = dst: structurally arange(E)//DEG (contiguous segments) -> not needed.

  int nNodes = in_sizes[0] / GAT_IN;

  char*  ws     = (char*)d_ws;
  float* h_ws   = (float*)ws;                                     // N*48 f32
  size_t hBytes = (size_t)nNodes * GAT_HC * sizeof(float);
  float* a1_ws  = (float*)(ws + hBytes);                          // N*4 f32
  size_t aBytes = (size_t)nNodes * GAT_H * sizeof(float);
  float* a2_ws  = (float*)(ws + hBytes + aBytes);                 // N*4 f32
  float* out    = (float*)d_out;

  int tiles   = (nNodes + 15) / 16;
  int blocks1 = (tiles + 7) / 8;          // 8 waves (16-node tiles) per 256-thread block
  gat_gemm_wmma<<<blocks1, 256, 0, stream>>>(x, wgt, h_ws, nNodes);

  int totalNH = nNodes * GAT_H;
  int blocks2 = (totalNH + 255) / 256;
  gat_logits<<<blocks2, 256, 0, stream>>>(h_ws, att_l, att_r, a1_ws, a2_ws, totalNH);
  gat_softmax_agg<<<blocks2, 256, 0, stream>>>(h_ws, a1_ws, a2_ws, src, out, totalNH);

  (void)n_in; (void)out_size; (void)ws_size;
}
